// GCN_4054449127728
// MI455X (gfx1250) — matrix operand order repaired and checked by reading, not compiled, not run on hardware
//
#include <hip/hip_runtime.h>
#include <hip/hip_bf16.h>

// ---------------------------------------------------------------------------
// GCN forward (5 GCNConv layers + linear head) for MI455X / gfx1250 (wave32).
// Dense h@W runs on v_wmma_f32_16x16x32_bf16 (bf16 in, fp32 accumulate).
// Edge gather/scatter stays fp32 with f32 atomics (L2-resident: hW is 25.6MB
// vs 192MB L2, so the random traffic never leaves L2).
// ---------------------------------------------------------------------------

#define N_NODES 50000
#define N_EDGES 1600000
#define IN_F    64
#define EMB     128

typedef __attribute__((ext_vector_type(4)))  __bf16 v4bf;
typedef __attribute__((ext_vector_type(8)))  __bf16 v8bf;
typedef __attribute__((ext_vector_type(16))) __bf16 v16bf;
typedef __attribute__((ext_vector_type(8)))  float  v8f;

// ---------------- graph normalization ----------------

__global__ void k_degree(const int* __restrict__ dst, float* __restrict__ deg) {
    int e = blockIdx.x * blockDim.x + threadIdx.x;
    if (e < N_EDGES) atomicAdd(&deg[dst[e]], 1.0f);
}

__global__ void k_norm(const float* __restrict__ deg,
                       float* __restrict__ dinv_sqrt,
                       float* __restrict__ deg_inv) {
    int i = blockIdx.x * blockDim.x + threadIdx.x;
    if (i < N_NODES) {
        float d = deg[i] + 1.0f;          // +1 self loop
        dinv_sqrt[i] = rsqrtf(d);
        deg_inv[i]   = 1.0f / d;
    }
}

__global__ void k_edge_norm(const int* __restrict__ src, const int* __restrict__ dst,
                            const float* __restrict__ dinv_sqrt,
                            float* __restrict__ en) {
    int e = blockIdx.x * blockDim.x + threadIdx.x;
    if (e < N_EDGES) en[e] = dinv_sqrt[src[e]] * dinv_sqrt[dst[e]];
}

// ---------------- precision conversion ----------------

__global__ void k_f32_to_bf16(const float* __restrict__ in, __bf16* __restrict__ out, int n) {
    int i = blockIdx.x * blockDim.x + threadIdx.x;
    if (i < n) out[i] = (__bf16)in[i];
}

// W [K x 128] f32 row-major  ->  Wt [128 x K] bf16 (column of W becomes a row)
__global__ void k_w_transpose(const float* __restrict__ W, __bf16* __restrict__ Wt, int K) {
    int i = blockIdx.x * blockDim.x + threadIdx.x;
    if (i < K * EMB) {
        int k = i / EMB, n = i % EMB;
        Wt[n * K + k] = (__bf16)W[i];
    }
}

// ---------------- WMMA GEMM: C[N x 128] = A[N x K] @ W[K x 128] ----------------
// One wave computes one 16x16 tile; 8 waves/block cover the full 128 output
// columns of one 16-row stripe. gridDim.x = N/16 = 3125 (exact).
template <int K>
__global__ void __launch_bounds__(256) k_gemm_wmma(const __bf16* __restrict__ A,
                                                   const __bf16* __restrict__ Wt,
                                                   float* __restrict__ C) {
    const int lane = threadIdx.x & 31;
    const int wave = threadIdx.x >> 5;   // output column tile, 0..7
    const int mt   = blockIdx.x;         // output row tile
    const int hi   = lane >> 4;          // half-wave select
    const int l15  = lane & 15;

    // A fragment (kernel's own k order, not the ISA table order): lane holds row M=l15,
    // K values [kk+16*hi, +16) contiguous, matching B.
    const __bf16* arow = A  + (size_t)(mt * 16 + l15) * K + hi * 16;
    // B fragment (same k order as A, not the ISA table order): lane holds column N=l15,
    // K values [kk+16*hi, +16) — contiguous in the transposed Wt.
    const __bf16* brow = Wt + (size_t)(wave * 16 + l15) * K + hi * 16;

    v8f c = {};
#pragma unroll
    for (int kk = 0; kk < K; kk += 32) {
        v16bf a = *(const v16bf*)(arow + kk);
        v16bf b = *(const v16bf*)(brow + kk);
        c = __builtin_amdgcn_wmma_f32_16x16x32_bf16(
                /*neg_a=*/false, a, /*neg_b=*/false, b,
                /*c_mod=*/(short)0, c, /*reuse_a=*/false, /*reuse_b=*/false);
    }

    // C/D layout: VGPR r -> row (r + 8*hi), column l15.
    float* out = C + (size_t)(mt * 16 + hi * 8) * EMB + wave * 16 + l15;
#pragma unroll
    for (int r = 0; r < 8; ++r) out[(size_t)r * EMB] = c[r];
}

// ---------------- edge gather * norm -> scatter-add ----------------
// One wave per edge; lane covers 4 contiguous features (128 = 32*4).
__global__ void __launch_bounds__(256) k_scatter(const int* __restrict__ src,
                                                 const int* __restrict__ dst,
                                                 const float* __restrict__ en,
                                                 const float* __restrict__ hW,
                                                 float* __restrict__ agg) {
    const int lane = threadIdx.x & 31;
    const int e    = blockIdx.x * 8 + (threadIdx.x >> 5);
    if (e >= N_EDGES) return;
    const int   s = src[e];
    const int   d = dst[e];
    const float w = en[e];
    const float4 v = *(const float4*)(hW + (size_t)s * EMB + lane * 4);
    float* ap = agg + (size_t)d * EMB + lane * 4;
    atomicAdd(ap + 0, v.x * w);
    atomicAdd(ap + 1, v.y * w);
    atomicAdd(ap + 2, v.z * w);
    atomicAdd(ap + 3, v.w * w);
}

// ---------------- self-loop + bias + ReLU (writes f32 and bf16 copies) ----------------
__global__ void k_finish(const float* __restrict__ agg, const float* __restrict__ hW,
                         const float* __restrict__ deg_inv, const float* __restrict__ bias,
                         float* __restrict__ h, __bf16* __restrict__ hbf) {
    int i = blockIdx.x * blockDim.x + threadIdx.x;   // float4 granularity
    if (i >= N_NODES * (EMB / 4)) return;
    int node = i >> 5;
    int c4   = (i & 31) * 4;
    float di = deg_inv[node];
    const float4 a  = *(const float4*)(agg + (size_t)node * EMB + c4);
    const float4 hw = *(const float4*)(hW  + (size_t)node * EMB + c4);
    const float4 b  = *(const float4*)(bias + c4);
    float4 r;
    r.x = fmaxf(fmaf(hw.x, di, a.x) + b.x, 0.0f);
    r.y = fmaxf(fmaf(hw.y, di, a.y) + b.y, 0.0f);
    r.z = fmaxf(fmaf(hw.z, di, a.z) + b.z, 0.0f);
    r.w = fmaxf(fmaf(hw.w, di, a.w) + b.w, 0.0f);
    *(float4*)(h + (size_t)node * EMB + c4) = r;
    v4bf rb; rb[0] = (__bf16)r.x; rb[1] = (__bf16)r.y; rb[2] = (__bf16)r.z; rb[3] = (__bf16)r.w;
    *(v4bf*)(hbf + (size_t)node * EMB + c4) = rb;
}

// ---------------- output head: out[n] = h[n,:] . Wout + bout ----------------
__global__ void __launch_bounds__(256) k_out(const float* __restrict__ h,
                                             const float* __restrict__ Wout,
                                             const float* __restrict__ bout,
                                             float* __restrict__ out) {
    const int lane = threadIdx.x & 31;
    const int n    = blockIdx.x * 8 + (threadIdx.x >> 5);
    if (n >= N_NODES) return;
    const float4 hv = *(const float4*)(h + (size_t)n * EMB + lane * 4);
    const float4 wv = *(const float4*)(Wout + lane * 4);
    float p = hv.x * wv.x + hv.y * wv.y + hv.z * wv.z + hv.w * wv.w;
#pragma unroll
    for (int off = 16; off > 0; off >>= 1) p += __shfl_down(p, off, 32);
    if (lane == 0) out[n] = p + bout[0];
}

// ---------------------------------------------------------------------------

static inline size_t align256(size_t x) { return (x + 255) & ~(size_t)255; }

extern "C" void kernel_launch(void* const* d_in, const int* in_sizes, int n_in,
                              void* d_out, int out_size, void* d_ws, size_t ws_size,
                              hipStream_t stream) {
    (void)in_sizes; (void)n_in; (void)out_size; (void)ws_size;

    // inputs (setup_inputs order)
    const float* x     = (const float*)d_in[1];
    const int*   eidx  = (const int*)d_in[2];
    const int*   src   = eidx;
    const int*   dst   = eidx + N_EDGES;
    const float* W[5]  = { (const float*)d_in[3], (const float*)d_in[5], (const float*)d_in[7],
                           (const float*)d_in[9], (const float*)d_in[11] };
    const float* B[5]  = { (const float*)d_in[4], (const float*)d_in[6], (const float*)d_in[8],
                           (const float*)d_in[10], (const float*)d_in[12] };
    const float* Wout  = (const float*)d_in[13];
    const float* bout  = (const float*)d_in[14];
    float*       out   = (float*)d_out;

    // workspace carve-up (256B aligned)
    char* p = (char*)d_ws;
    float*  deg  = (float*)p;  p += align256((size_t)N_NODES * 4);
    float*  dis  = (float*)p;  p += align256((size_t)N_NODES * 4);   // deg^-1/2
    float*  dinv = (float*)p;  p += align256((size_t)N_NODES * 4);   // deg^-1
    float*  en   = (float*)p;  p += align256((size_t)N_EDGES * 4);
    float*  hW   = (float*)p;  p += align256((size_t)N_NODES * EMB * 4);
    float*  agg  = (float*)p;  p += align256((size_t)N_NODES * EMB * 4);
    float*  h    = (float*)p;  p += align256((size_t)N_NODES * EMB * 4);
    __bf16* hbf  = (__bf16*)p; p += align256((size_t)N_NODES * EMB * 2);
    __bf16* Wt   = (__bf16*)p; p += align256((size_t)EMB * EMB * 2);

    const int T = 256;

    // ---- normalization (one time) ----
    hipMemsetAsync(deg, 0, (size_t)N_NODES * 4, stream);
    k_degree<<<(N_EDGES + T - 1) / T, T, 0, stream>>>(dst, deg);
    k_norm<<<(N_NODES + T - 1) / T, T, 0, stream>>>(deg, dis, dinv);
    k_edge_norm<<<(N_EDGES + T - 1) / T, T, 0, stream>>>(src, dst, dis, en);

    // ---- layer 0 input: x -> bf16 (N x 64) ----
    k_f32_to_bf16<<<(N_NODES * IN_F + T - 1) / T, T, 0, stream>>>(x, hbf, N_NODES * IN_F);

    for (int layer = 0; layer < 5; ++layer) {
        const int K = (layer == 0) ? IN_F : EMB;
        k_w_transpose<<<(K * EMB + T - 1) / T, T, 0, stream>>>(W[layer], Wt, K);
        if (K == IN_F)
            k_gemm_wmma<IN_F><<<N_NODES / 16, T, 0, stream>>>(hbf, Wt, hW);
        else
            k_gemm_wmma<EMB><<<N_NODES / 16, T, 0, stream>>>(hbf, Wt, hW);
        hipMemsetAsync(agg, 0, (size_t)N_NODES * EMB * 4, stream);
        k_scatter<<<N_EDGES / 8, T, 0, stream>>>(src, dst, en, hW, agg);
        k_finish<<<(N_NODES * (EMB / 4) + T - 1) / T, T, 0, stream>>>(agg, hW, dinv, B[layer], h, hbf);
    }

    k_out<<<N_NODES / 8, T, 0, stream>>>(h, Wout, bout, out);
}
